// OneDeltaPerBinModel_23089744183947
// MI455X (gfx1250) — matrix operand order repaired and checked by reading, not compiled, not run on hardware
//
#include <hip/hip_runtime.h>
#include <hip/hip_bf16.h>

typedef __attribute__((ext_vector_type(16))) __bf16 v16bf;
typedef __attribute__((ext_vector_type(8)))  float  v8f;
typedef unsigned short u16;

// ---------------- problem constants ----------------
#define BATCH 4096
#define N0    2048
#define NCLS  100
#define DIN   2148          // N0 + NCLS
#define DPAD  2176          // DIN padded to multiple of 32
#define N1R   1000
#define N1P   1024
#define N2R   500
#define N2P   512
#define KR    200           // num clusters
#define KP    256           // padded for GEMM3 tile grid
#define N3    100
#define N3P   112           // N3 padded to multiple of 16
#define NDIM  3
#define GDIM  4             // row-group blocks per cluster (stride loop covers any count)

__device__ __forceinline__ u16 f2bf(float f) {
    union { float f; unsigned u; } v; v.f = f;
    unsigned u = v.u;
    u += 0x7fffu + ((u >> 16) & 1u);   // round to nearest even
    return (u16)(u >> 16);
}

// ---------------- prep: xin = [x | onehot(label)] in bf16, zero padded ----------------
__global__ void build_xin(const float* __restrict__ x, const int* __restrict__ label,
                          u16* __restrict__ xin) {
    size_t i = (size_t)blockIdx.x * blockDim.x + threadIdx.x;
    if (i >= (size_t)BATCH * DPAD) return;
    int b = (int)(i / DPAD);
    int d = (int)(i % DPAD);
    float v;
    if (d < N0)        v = x[(size_t)b * N0 + d];
    else if (d < DIN)  v = (d - N0 == label[b]) ? 1.f : 0.f;
    else               v = 0.f;
    xin[i] = f2bf(v);
}

// ---------------- prep: LDS-tiled transpose fp32 -> bf16 ----------------
// dst[c][r] = src[r][c]; src: R x C fp32 row-major (+k*srcK), dst: Cpad x Rpad bf16 (+k*dstK).
// grid: (ceil(Rpad/32), ceil(Cpad/32), nk), block: (32, 8)
__global__ void transpose_bf16_tiled(const float* __restrict__ src, u16* __restrict__ dst,
                                     int R, int C, int Rpad, int Cpad,
                                     size_t srcK, size_t dstK) {
    __shared__ float t[32][33];
    const float* s = src + (size_t)blockIdx.z * srcK;
    u16* d = dst + (size_t)blockIdx.z * dstK;
    const int r0 = blockIdx.x * 32;
    const int c0 = blockIdx.y * 32;
    const int tx = threadIdx.x, ty = threadIdx.y;
#pragma unroll
    for (int j = 0; j < 4; ++j) {
        int r = r0 + ty + j * 8;
        int c = c0 + tx;
        t[ty + j * 8][tx] = (r < R && c < C) ? s[(size_t)r * C + c] : 0.f;
    }
    __syncthreads();
#pragma unroll
    for (int j = 0; j < 4; ++j) {
        int c = c0 + ty + j * 8;
        int r = r0 + tx;
        if (c < Cpad && r < Rpad) d[(size_t)c * Rpad + r] = f2bf(t[tx][ty + j * 8]);
    }
}

// ---------------- WMMA fragment loaders (ISA 7.12.2 bf16 layouts) ----------------
union Frag { v16bf v; uint4 q[2]; };

// A: 16x32 bf16 row-major, base pre-offset to (row0, k0).
__device__ __forceinline__ v16bf load_a_frag(const u16* __restrict__ A, int lda, int lane) {
    Frag f;
    const u16* p = A + (size_t)(lane & 15) * lda + ((lane >> 4) << 3);
    f.q[0] = *(const uint4*)p;
    f.q[1] = *(const uint4*)(p + 16);
    return f.v;
}

// A fragment with per-lane row pointer (row-gathered GEMM). prow = &xin[rowIdx * lda].
__device__ __forceinline__ v16bf load_a_frag_gather(const u16* __restrict__ prow, int k0, int lane) {
    Frag f;
    const u16* p = prow + k0 + ((lane >> 4) << 3);
    f.q[0] = *(const uint4*)p;
    f.q[1] = *(const uint4*)(p + 16);
    return f.v;
}

// B: 32x16 bf16 from Bt (N x K row-major), base pre-offset to (col0, k0).
__device__ __forceinline__ v16bf load_b_frag(const u16* __restrict__ Bt, int ldb, int lane) {
    Frag f;
    const u16* p = Bt + (size_t)(lane & 15) * ldb + ((lane >> 4) << 3);
    f.q[0] = *(const uint4*)p;
    f.q[1] = *(const uint4*)(p + 16);
    return f.v;
}

__device__ __forceinline__ v8f wmma_bf16(v16bf a, v16bf b, v8f c) {
    return __builtin_amdgcn_wmma_f32_16x16x32_bf16(false, a, false, b, (short)0, c, false, false);
}

// ---------------- bf16 WMMA GEMM: one wave computes a 32x64 C tile ----------------
template <bool RELU_BF16_OUT>
__global__ void gemm_wmma(const u16* __restrict__ A, int lda,
                          const u16* __restrict__ Bt, int ldb,
                          const float* __restrict__ bias, int Nreal, int Kk,
                          u16* __restrict__ outB, float* __restrict__ outF, int ldo) {
    const int lane = threadIdx.x & 31;
    const int row0 = blockIdx.y * 32;
    const int col0 = blockIdx.x * 64;

    v8f c[2][4];
#pragma unroll
    for (int i = 0; i < 2; ++i)
#pragma unroll
        for (int j = 0; j < 4; ++j) c[i][j] = (v8f)(0.0f);

    const u16* Ab = A + (size_t)row0 * lda;
    const u16* Bb = Bt + (size_t)col0 * ldb;

    for (int k0 = 0; k0 < Kk; k0 += 32) {
        v16bf a0 = load_a_frag(Ab + k0,                    lda, lane);
        v16bf a1 = load_a_frag(Ab + (size_t)16 * lda + k0, lda, lane);
        v16bf b0 = load_b_frag(Bb + k0,                    ldb, lane);
        v16bf b1 = load_b_frag(Bb + (size_t)16 * ldb + k0, ldb, lane);
        v16bf b2 = load_b_frag(Bb + (size_t)32 * ldb + k0, ldb, lane);
        v16bf b3 = load_b_frag(Bb + (size_t)48 * ldb + k0, ldb, lane);
        c[0][0] = wmma_bf16(a0, b0, c[0][0]);
        c[0][1] = wmma_bf16(a0, b1, c[0][1]);
        c[0][2] = wmma_bf16(a0, b2, c[0][2]);
        c[0][3] = wmma_bf16(a0, b3, c[0][3]);
        c[1][0] = wmma_bf16(a1, b0, c[1][0]);
        c[1][1] = wmma_bf16(a1, b1, c[1][1]);
        c[1][2] = wmma_bf16(a1, b2, c[1][2]);
        c[1][3] = wmma_bf16(a1, b3, c[1][3]);
    }

    const int nsub = lane & 15;
    const int radd = (lane >> 4) << 3;
#pragma unroll
    for (int mt = 0; mt < 2; ++mt)
#pragma unroll
        for (int nt = 0; nt < 4; ++nt)
#pragma unroll
            for (int r = 0; r < 8; ++r) {
                int mrow = row0 + mt * 16 + r + radd;
                int ncol = col0 + nt * 16 + nsub;
                float bv = (ncol < Nreal) ? bias[ncol] : 0.f;
                float v  = c[mt][nt][r] + bv;
                if (RELU_BF16_OUT) {
                    v = fmaxf(v, 0.f);
                    outB[(size_t)mrow * ldo + ncol] = f2bf(v);
                } else {
                    if (ncol < Nreal) outF[(size_t)mrow * ldo + ncol] = v;
                }
            }
}

// ---------------- argmax over K=200 logits, one thread per row ----------------
__global__ void argmax_rows(const float* __restrict__ y1, int* __restrict__ kstar) {
    int b = blockIdx.x * blockDim.x + threadIdx.x;
    if (b >= BATCH) return;
    const float* r = y1 + (size_t)b * KR;
    float best = r[0];
    int bi = 0;
    for (int j = 1; j < KR; ++j) {
        float v = r[j];
        if (v > best) { best = v; bi = j; }
    }
    kstar[b] = bi;
}

// ---------------- cluster grouping: zero / histogram / scan / scatter ----------------
__global__ void zero_counts(int* __restrict__ counts) {
    int i = threadIdx.x;
    if (i < KR) counts[i] = 0;
}
__global__ void hist_clusters(const int* __restrict__ kstar, int* __restrict__ counts) {
    int b = blockIdx.x * blockDim.x + threadIdx.x;
    if (b < BATCH) atomicAdd(&counts[kstar[b]], 1);
}
__global__ void scan_counts(const int* __restrict__ counts, int* __restrict__ offsets,
                            int* __restrict__ cursor) {
    if (threadIdx.x == 0) {
        int s = 0;
        for (int k = 0; k < KR; ++k) { offsets[k] = s; cursor[k] = s; s += counts[k]; }
    }
}
__global__ void scatter_rows(const int* __restrict__ kstar, int* __restrict__ cursor,
                             int* __restrict__ rowlist) {
    int b = blockIdx.x * blockDim.x + threadIdx.x;
    if (b >= BATCH) return;
    int k = kstar[b];
    int pos = atomicAdd(&cursor[k], 1);
    rowlist[pos] = b;
}

// ---------------- grouped res_2layer: one wave per (cluster, 16-row group) ----------------
// WMMA GEMM: A = gathered xin rows [16 x 2176] bf16, B = Wr1[k]^T [112 x 2176] bf16.
// Epilogue: +br1, relu -> LDS, then the 100x3 second layer as 48 VALU dots.
__global__ void res_grouped(const u16* __restrict__ xin, const u16* __restrict__ WrT1,
                            const float* __restrict__ br1,
                            const float* __restrict__ Wr2, const float* __restrict__ br2,
                            const int* __restrict__ counts, const int* __restrict__ offsets,
                            const int* __restrict__ rowlist, float* __restrict__ y2) {
    __shared__ float hs[16][N3P + 1];
    __shared__ int   rloc[16];
    const int k    = blockIdx.x;
    const int g    = blockIdx.y;
    const int lane = threadIdx.x & 31;
    const int cnt  = counts[k];
    const int base = offsets[k];
    const u16* Bk  = WrT1 + (size_t)k * N3P * DPAD;

    for (int rbase = g * 16; rbase < cnt; rbase += 16 * GDIM) {
        if (lane < 16) {
            int idx = rbase + lane;
            rloc[lane] = rowlist[base + ((idx < cnt) ? idx : rbase)];
        }
        __syncthreads();
        const u16* Arow = xin + (size_t)rloc[lane & 15] * DPAD;

        v8f c[7];
#pragma unroll
        for (int nt = 0; nt < 7; ++nt) c[nt] = (v8f)(0.0f);

        for (int k0 = 0; k0 < DPAD; k0 += 32) {
            v16bf a = load_a_frag_gather(Arow, k0, lane);
#pragma unroll
            for (int nt = 0; nt < 7; ++nt) {
                v16bf b = load_b_frag(Bk + (size_t)nt * 16 * DPAD + k0, DPAD, lane);
                c[nt] = wmma_bf16(a, b, c[nt]);
            }
        }

        const int nsub = lane & 15;
        const int radd = (lane >> 4) << 3;
#pragma unroll
        for (int nt = 0; nt < 7; ++nt) {
            int n = nt * 16 + nsub;
#pragma unroll
            for (int r = 0; r < 8; ++r) {
                int m = r + radd;
                if (n < N3) hs[m][n] = fmaxf(c[nt][r] + br1[k * N3 + n], 0.f);
            }
        }
        __syncthreads();

        // second layer: y2[row] = hs[row] . Wr2[k] + br2[k]   (48 dot products, len 100)
        for (int t = lane; t < 16 * NDIM; t += 32) {
            int j = t / NDIM;
            int m = t % NDIM;
            if (rbase + j < cnt) {
                int row = rloc[j];
                const float* w2 = Wr2 + (size_t)k * N3 * NDIM + m;
                float acc = br2[k * NDIM + m];
#pragma unroll 4
                for (int n = 0; n < N3; ++n) acc = fmaf(hs[j][n], w2[(size_t)n * NDIM], acc);
                y2[(size_t)row * NDIM + m] = acc;
            }
        }
        __syncthreads();
    }
}

// ---------------- host launcher ----------------
extern "C" void kernel_launch(void* const* d_in, const int* in_sizes, int n_in,
                              void* d_out, int out_size, void* d_ws, size_t ws_size,
                              hipStream_t stream) {
    const float* x   = (const float*)d_in[0];
    const int*   lab = (const int*)  d_in[1];
    const float* Wb1 = (const float*)d_in[2];
    const float* bb1 = (const float*)d_in[3];
    const float* Wb2 = (const float*)d_in[4];
    const float* bb2 = (const float*)d_in[5];
    const float* Wb3 = (const float*)d_in[6];
    const float* bb3 = (const float*)d_in[7];
    const float* Wr1 = (const float*)d_in[8];
    const float* br1 = (const float*)d_in[9];
    const float* Wr2 = (const float*)d_in[10];
    const float* br2 = (const float*)d_in[11];

    float* y1 = (float*)d_out;                 // [4096 x 200]
    float* y2 = y1 + (size_t)BATCH * KR;       // [4096 x 3]

    char* ws = (char*)d_ws;
    auto carve = [&](size_t bytes) {
        char* p = ws;
        ws += (bytes + 255) & ~(size_t)255;
        return p;
    };
    u16* xin   = (u16*)carve((size_t)BATCH * DPAD * 2);        // 17.8 MB
    u16* WbT1  = (u16*)carve((size_t)N1P * DPAD * 2);          //  4.5 MB
    u16* WbT2  = (u16*)carve((size_t)N2P * N1P * 2);           //  1.0 MB
    u16* WbT3  = (u16*)carve((size_t)KP * N2P * 2);            //  0.3 MB
    u16* WrT1  = (u16*)carve((size_t)KR * N3P * DPAD * 2);     // 97.5 MB
    u16* h1    = (u16*)carve((size_t)BATCH * N1P * 2);         //  8.4 MB
    u16* h2    = (u16*)carve((size_t)BATCH * N2P * 2);         //  4.2 MB
    int* kst   = (int*)carve((size_t)BATCH * 4);
    int* rows  = (int*)carve((size_t)BATCH * 4);
    int* cnts  = (int*)carve((size_t)KR * 4);
    int* offs  = (int*)carve((size_t)KR * 4);
    int* curs  = (int*)carve((size_t)KR * 4);

    // ---- prep: bf16 conversions / tiled transposes ----
    {
        size_t n = (size_t)BATCH * DPAD;
        build_xin<<<dim3((unsigned)((n + 255) / 256)), dim3(256), 0, stream>>>(x, lab, xin);
    }
    dim3 tb(32, 8);
    // WbT1[n][d] = Wb1[d][n] : src 2148x1000 -> dst 1024x2176
    transpose_bf16_tiled<<<dim3(DPAD / 32, N1P / 32, 1), tb, 0, stream>>>(
        Wb1, WbT1, DIN, N1R, DPAD, N1P, 0, 0);
    // WbT2[n2][n1] = Wb2[n1][n2] : src 1000x500 -> dst 512x1024
    transpose_bf16_tiled<<<dim3(N1P / 32, N2P / 32, 1), tb, 0, stream>>>(
        Wb2, WbT2, N1R, N2R, N1P, N2P, 0, 0);
    // WbT3[k][n2] = Wb3[n2][k] : src 500x200 -> dst 256x512
    transpose_bf16_tiled<<<dim3(N2P / 32, KP / 32, 1), tb, 0, stream>>>(
        Wb3, WbT3, N2R, KR, N2P, KP, 0, 0);
    // WrT1[k][n][d] = Wr1[k][d][n] : 200 slices of 2148x100 -> 112x2176
    transpose_bf16_tiled<<<dim3(DPAD / 32, (N3P + 31) / 32, KR), tb, 0, stream>>>(
        Wr1, WrT1, DIN, N3, DPAD, N3P, (size_t)DIN * N3, (size_t)N3P * DPAD);

    // ---- bin_3layer GEMMs (bf16 WMMA, f32 accumulate) ----
    gemm_wmma<true><<<dim3(N1P / 64, BATCH / 32), dim3(32), 0, stream>>>(
        xin, DPAD, WbT1, DPAD, bb1, N1R, DPAD, h1, nullptr, N1P);
    gemm_wmma<true><<<dim3(N2P / 64, BATCH / 32), dim3(32), 0, stream>>>(
        h1, N1P, WbT2, N1P, bb2, N2R, N1P, h2, nullptr, N2P);
    gemm_wmma<false><<<dim3(KP / 64, BATCH / 32), dim3(32), 0, stream>>>(
        h2, N2P, WbT3, N2P, bb3, KR, N2P, nullptr, y1, KR);

    // ---- argmax -> kstar, then group rows by cluster ----
    argmax_rows<<<dim3((BATCH + 255) / 256), dim3(256), 0, stream>>>(y1, kst);
    zero_counts<<<dim3(1), dim3(256), 0, stream>>>(cnts);
    hist_clusters<<<dim3(BATCH / 256), dim3(256), 0, stream>>>(kst, cnts);
    scan_counts<<<dim3(1), dim3(32), 0, stream>>>(cnts, offs, curs);
    scatter_rows<<<dim3(BATCH / 256), dim3(256), 0, stream>>>(kst, curs, rows);

    // ---- grouped per-cluster res_2layer (WMMA, Wr1^T streamed once per group) ----
    res_grouped<<<dim3(KR, GDIM), dim3(32), 0, stream>>>(
        xin, WrT1, br1, Wr2, br2, cnts, offs, rows, y2);
}
